// CrossAttention_80676665688225
// MI455X (gfx1250) — matrix operand order repaired and checked by reading, not compiled, not run on hardware
//
#include <hip/hip_runtime.h>

// ---------------------------------------------------------------------------
// CDNA5 (gfx1250) fused cross-attention.
// All GEMMs via v_wmma_f32_16x16x32_bf16; K/V tiles staged to LDS with
// global_load_async_to_lds_b128 (ASYNCcnt) double-buffered; base-2 softmax.
// B=8, C=D=128, N=4096 (x tokens), M=1024 (y tokens), wave32.
// ---------------------------------------------------------------------------

typedef __attribute__((__ext_vector_type__(16))) __bf16 v16bf;
typedef __attribute__((__ext_vector_type__(8)))  float  v8f;

union V16 {
  v16bf v;
  uint4 u[2];
};

__device__ __forceinline__ v8f v8zero() {
  v8f z;
#pragma unroll
  for (int i = 0; i < 8; ++i) z[i] = 0.0f;
  return z;
}

// Load 16 contiguous bf16 (B-operand: lane supplies K = 16h .. 16h+15).
// Works for both global and LDS pointers (addrspace inferred).
__device__ __forceinline__ v16bf ld16(const __bf16* p) {
  V16 t;
  t.u[0] = *(const uint4*)(p);
  t.u[1] = *(const uint4*)(p + 8);
  return t.v;
}

__device__ __forceinline__ v8f wmma_bf16(v16bf a, v16bf b, v8f c) {
  // (neg_a, A, neg_b, B, c_mod, C, reuse_a, reuse_b)
  return __builtin_amdgcn_wmma_f32_16x16x32_bf16(false, a, false, b, (short)0, c,
                                                 false, false);
}

// Async copy of 16 bytes global -> LDS, tracked by ASYNCcnt.
__device__ __forceinline__ void async_b128(const __bf16* gptr, __bf16* lptr) {
  unsigned lo = (unsigned)(uintptr_t)lptr;      // low 32 bits = LDS offset
  unsigned long long ga = (unsigned long long)(uintptr_t)gptr;
  asm volatile("global_load_async_to_lds_b128 %0, %1, off"
               :: "v"(lo), "v"(ga) : "memory");
}

// ---------------------------------------------------------------------------
// Kernel 0: convert weights f32 -> bf16 (row-major (d,c): exactly the
// B-operand layout for out[n,d] = sum_c X[n,c]*W[d,c]).
// ---------------------------------------------------------------------------
__global__ __launch_bounds__(256) void k_convw(
    const float* __restrict__ Wq, const float* __restrict__ Wk,
    const float* __restrict__ Wv, __bf16* __restrict__ wq,
    __bf16* __restrict__ wk, __bf16* __restrict__ wv) {
  int i = blockIdx.x * 256 + threadIdx.x;
  if (i < 128 * 128) {
    wq[i] = (__bf16)Wq[i];
    wk[i] = (__bf16)Wk[i];
    wv[i] = (__bf16)Wv[i];
  }
}

// Stage 64 tokens x 128 channels from (B,C,NTOK) f32 into LDS [64][136] bf16.
template <int NTOK>
__device__ __forceinline__ void stage_tokens(const float* __restrict__ src,
                                             int b, int n0,
                                             __bf16 (*xt)[136]) {
  int t = threadIdx.x;
  int c = t >> 1;
  int h2 = t & 1;
#pragma unroll
  for (int cc = 0; cc < 2; ++cc) {
    int ch = c + 64 * cc;
    const float* s = src + ((size_t)b * 128 + ch) * NTOK + n0 + 32 * h2;
#pragma unroll
    for (int k = 0; k < 32; k += 4) {
      float4 f = *(const float4*)(s + k);
      xt[32 * h2 + k + 0][ch] = (__bf16)f.x;
      xt[32 * h2 + k + 1][ch] = (__bf16)f.y;
      xt[32 * h2 + k + 2][ch] = (__bf16)f.z;
      xt[32 * h2 + k + 3][ch] = (__bf16)f.w;
    }
  }
}

// A fragment (16 rows x K=32 bf16) from a K-contiguous row pointer.
// elements [0..15] = K[16h..16h+15]
__device__ __forceinline__ v16bf ldA(const __bf16* r, int h) {
  V16 t;
  t.u[0] = *(const uint4*)(r + 16 * h);
  t.u[1] = *(const uint4*)(r + 16 * h + 8);
  return t.v;
}

// ---------------------------------------------------------------------------
// Kernel 1: Q projection, scale = log2(e)/sqrt(128) folded in (base-2 softmax)
// Qb[b][n][d], d contiguous. grid = 512 blocks, 128 threads.
// ---------------------------------------------------------------------------
__global__ __launch_bounds__(128) void k_proj_q(
    const float* __restrict__ x, const __bf16* __restrict__ Wqb,
    const float* __restrict__ bq, __bf16* __restrict__ Qb) {
  __shared__ __bf16 xt[64][136];
  int b = blockIdx.x >> 6;
  int n0 = (blockIdx.x & 63) << 6;
  stage_tokens<4096>(x, b, n0, xt);
  __syncthreads();

  int t = threadIdx.x, w = t >> 5, lane = t & 31, col = lane & 15, h = lane >> 4;
  v16bf a[4];
#pragma unroll
  for (int cc = 0; cc < 4; ++cc) a[cc] = ldA(&xt[16 * w + col][32 * cc], h);

  const float scale = 0.12751744f;  // (1/sqrt(128)) * log2(e)
  for (int dt = 0; dt < 8; ++dt) {
    int d0 = dt * 16;
    v8f acc = v8zero();
#pragma unroll
    for (int cc = 0; cc < 4; ++cc)
      acc = wmma_bf16(a[cc],
                      ld16(Wqb + (size_t)(d0 + col) * 128 + 32 * cc + 16 * h),
                      acc);
    float bias = bq[d0 + col];
    __bf16* dst = Qb + ((size_t)b * 4096 + n0 + 16 * w) * 128 + d0 + col;
#pragma unroll
    for (int i = 0; i < 8; ++i)
      dst[(size_t)(i + 8 * h) * 128] = (__bf16)((acc[i] + bias) * scale);
  }
}

// ---------------------------------------------------------------------------
// Kernel 2: K and V projection from one y staging.
//   Kb[b][m][d]  (d contiguous: B-operand of Q.K^T)
//   Vt[b][d][m]  (m contiguous: B-operand of P.V)
// grid = 128 blocks, 128 threads.
// ---------------------------------------------------------------------------
__global__ __launch_bounds__(128) void k_proj_kv(
    const float* __restrict__ y, const __bf16* __restrict__ Wkb,
    const float* __restrict__ bk, const __bf16* __restrict__ Wvb,
    const float* __restrict__ bv, __bf16* __restrict__ Kb,
    __bf16* __restrict__ Vt) {
  __shared__ __bf16 yt[64][136];
  int b = blockIdx.x >> 4;
  int m0 = (blockIdx.x & 15) << 6;
  stage_tokens<1024>(y, b, m0, yt);
  __syncthreads();

  int t = threadIdx.x, w = t >> 5, lane = t & 31, col = lane & 15, h = lane >> 4;
  v16bf a[4];
#pragma unroll
  for (int cc = 0; cc < 4; ++cc) a[cc] = ldA(&yt[16 * w + col][32 * cc], h);

  for (int dt = 0; dt < 8; ++dt) {
    int d0 = dt * 16;
    v8f ak = v8zero(), av = v8zero();
#pragma unroll
    for (int cc = 0; cc < 4; ++cc) {
      ak = wmma_bf16(a[cc],
                     ld16(Wkb + (size_t)(d0 + col) * 128 + 32 * cc + 16 * h),
                     ak);
      av = wmma_bf16(a[cc],
                     ld16(Wvb + (size_t)(d0 + col) * 128 + 32 * cc + 16 * h),
                     av);
    }
    float bkv = bk[d0 + col];
    float bvv = bv[d0 + col];
    __bf16* kd = Kb + ((size_t)b * 1024 + m0 + 16 * w) * 128 + d0 + col;
    __bf16* vd = Vt + ((size_t)b * 128 + d0 + col) * 1024 + m0 + 16 * w;
#pragma unroll
    for (int i = 0; i < 8; ++i) {
      kd[(size_t)(i + 8 * h) * 128] = (__bf16)(ak[i] + bkv);
      vd[i + 8 * h] = (__bf16)(av[i] + bvv);
    }
  }
}

// ---------------------------------------------------------------------------
// Kernel 3: flash attention + residual.
// grid = 512 blocks, 128 threads (4 waves, 16 Q rows each).
// K/V tiles (8 KB each) staged once per workgroup via async-to-LDS, double
// buffered; per m-step of 32: 8 WMMAs for S, base-2 online softmax with
// wave32 xor-shuffle row reductions, P transposed through padded per-wave
// LDS into an A fragment, 8 WMMAs for P.V.
// ---------------------------------------------------------------------------
#define KP 136  // K LDS pitch (bf16): 32 rows(m) x 128(d) + pad
#define VP 40   // V LDS pitch (bf16): 128 rows(d) x 32(m) + pad

__device__ __forceinline__ void issue_tile(const __bf16* kg, const __bf16* vg,
                                           __bf16* kl, __bf16* vl, int t) {
  // K tile: 32 x 128 bf16 = 512 x 16B chunks; 4 per thread, 8 issues/wave.
#pragma unroll
  for (int j = 0; j < 4; ++j) {
    int ch = t + 128 * j;
    int row = ch >> 4;
    int co = (ch & 15) * 8;
    async_b128(kg + (size_t)row * 128 + co, kl + row * KP + co);
  }
  // V tile: 128 x 32 bf16 = 512 x 16B chunks.
#pragma unroll
  for (int j = 0; j < 4; ++j) {
    int ch = t + 128 * j;
    int d = ch >> 2;
    int mo = (ch & 3) * 8;
    async_b128(vg + (size_t)d * 1024 + mo, vl + d * VP + mo);
  }
}

__global__ __launch_bounds__(128) void k_attn(
    const float* __restrict__ x, const __bf16* __restrict__ Qb,
    const __bf16* __restrict__ Kb, const __bf16* __restrict__ Vt,
    float* __restrict__ out) {
  __shared__ __bf16 Kl[2][32 * KP];   // 2 x 8704 B
  __shared__ __bf16 Vl[2][128 * VP];  // 2 x 10240 B
  __shared__ float pbuf[4][16 * 34];  // per-wave P transpose
  __shared__ float obuf[4][16 * 17];  // per-wave O staging

  int b = blockIdx.x >> 6;
  int n0 = (blockIdx.x & 63) << 6;
  int t = threadIdx.x, w = t >> 5, lane = t & 31, col = lane & 15, h = lane >> 4;
  int nw = n0 + 16 * w;

  const __bf16* kbase = Kb + (size_t)b * 1024 * 128;
  const __bf16* vbase = Vt + (size_t)b * 128 * 1024;

  // Kick off tile 0 copy, overlap with Q fragment loads.
  issue_tile(kbase, vbase, Kl[0], Vl[0], t);

  const __bf16* qrow = Qb + ((size_t)b * 4096 + nw + col) * 128;
  v16bf qf[4];
#pragma unroll
  for (int cc = 0; cc < 4; ++cc) qf[cc] = ldA(qrow + 32 * cc, h);

  v8f of[8];
#pragma unroll
  for (int i = 0; i < 8; ++i) of[i] = v8zero();
  float mrow[8], lrow[8];
#pragma unroll
  for (int i = 0; i < 8; ++i) {
    mrow[i] = -3.0e38f;
    lrow[i] = 0.0f;
  }

  float* P = pbuf[w];
  int cur = 0;

  for (int m0 = 0; m0 < 1024; m0 += 32) {
    if (m0 + 32 < 1024) {
      issue_tile(kbase + (size_t)(m0 + 32) * 128, vbase + (m0 + 32),
                 Kl[cur ^ 1], Vl[cur ^ 1], t);
      asm volatile("s_wait_asynccnt 0x8" ::: "memory");  // current tile landed
    } else {
      asm volatile("s_wait_asynccnt 0x0" ::: "memory");
    }
    __syncthreads();  // tile visible to all waves

    const __bf16* kl = Kl[cur];
    const __bf16* vl = Vl[cur];

    v8f s0 = v8zero(), s1 = v8zero();
    const __bf16* kr0 = kl + (size_t)col * KP + 16 * h;         // m-local = col
    const __bf16* kr1 = kl + (size_t)(16 + col) * KP + 16 * h;  // m-local+16
#pragma unroll
    for (int cc = 0; cc < 4; ++cc) {
      s0 = wmma_bf16(qf[cc], ld16(kr0 + 32 * cc), s0);
      s1 = wmma_bf16(qf[cc], ld16(kr1 + 32 * cc), s1);
    }

    // Base-2 online softmax (log2e/sqrt(128) folded into Q).
    float corr[8];
#pragma unroll
    for (int i = 0; i < 8; ++i) {
      float tmax = fmaxf(s0[i], s1[i]);
      tmax = fmaxf(tmax, __shfl_xor(tmax, 1, 32));
      tmax = fmaxf(tmax, __shfl_xor(tmax, 2, 32));
      tmax = fmaxf(tmax, __shfl_xor(tmax, 4, 32));
      tmax = fmaxf(tmax, __shfl_xor(tmax, 8, 32));
      float mnew = fmaxf(mrow[i], tmax);
      float c = __builtin_amdgcn_exp2f(mrow[i] - mnew);
      float p0 = __builtin_amdgcn_exp2f(s0[i] - mnew);
      float p1 = __builtin_amdgcn_exp2f(s1[i] - mnew);
      float rs = p0 + p1;
      rs += __shfl_xor(rs, 1, 32);
      rs += __shfl_xor(rs, 2, 32);
      rs += __shfl_xor(rs, 4, 32);
      rs += __shfl_xor(rs, 8, 32);
      mrow[i] = mnew;
      lrow[i] = lrow[i] * c + rs;
      corr[i] = c;
      P[(i + 8 * h) * 34 + col] = p0;       // (row n-local, col m-local)
      P[(i + 8 * h) * 34 + 16 + col] = p1;  // second 16-col tile
    }
#pragma unroll
    for (int tt = 0; tt < 8; ++tt) {
#pragma unroll
      for (int i = 0; i < 8; ++i) of[tt][i] *= corr[i];
    }

    // Transpose P (C/D layout -> A layout) through per-wave LDS.
    V16 pf;
    const float* prow = &P[col * 34 + 16 * h];
#pragma unroll
    for (int e = 0; e < 8; ++e) {
      pf.v[e] = (__bf16)prow[e];          // K = 16h + e
      pf.v[8 + e] = (__bf16)prow[8 + e];  // K = 16h + 8 + e
    }

    // O += P * V : 8 d-tiles of 16, B operand from LDS (m contiguous).
#pragma unroll
    for (int tt = 0; tt < 8; ++tt) {
      of[tt] = wmma_bf16(pf.v, ld16(vl + (size_t)(16 * tt + col) * VP + 16 * h),
                         of[tt]);
    }

    __syncthreads();  // all waves done with tile `cur` before it is re-filled
    cur ^= 1;
  }

  // Epilogue: z = O/l, out[b][d][n] = x + z, LDS transpose for coalescing.
  float rl[8];
#pragma unroll
  for (int i = 0; i < 8; ++i) rl[i] = 1.0f / lrow[i];
  float* O = obuf[w];
  int drow = lane >> 1;
  int nh = (lane & 1) * 8;
  for (int tt = 0; tt < 8; ++tt) {
#pragma unroll
    for (int i = 0; i < 8; ++i)
      O[col * 17 + i + 8 * h] = of[tt][i] * rl[i];  // O[d-local][n-local]
    const float* xs = x + ((size_t)b * 128 + 16 * tt + drow) * 4096 + nw + nh;
    float* os = out + ((size_t)b * 128 + 16 * tt + drow) * 4096 + nw + nh;
    const float* orow = &O[drow * 17 + nh];
    float4 x0 = *(const float4*)(xs);
    float4 x1 = *(const float4*)(xs + 4);
    float4 r0, r1;
    r0.x = x0.x + orow[0]; r0.y = x0.y + orow[1];
    r0.z = x0.z + orow[2]; r0.w = x0.w + orow[3];
    r1.x = x1.x + orow[4]; r1.y = x1.y + orow[5];
    r1.z = x1.z + orow[6]; r1.w = x1.w + orow[7];
    *(float4*)(os) = r0;
    *(float4*)(os + 4) = r1;
  }
}

// ---------------------------------------------------------------------------
// Launch
// ---------------------------------------------------------------------------
extern "C" void kernel_launch(void* const* d_in, const int* in_sizes, int n_in,
                              void* d_out, int out_size, void* d_ws,
                              size_t ws_size, hipStream_t stream) {
  (void)in_sizes; (void)n_in; (void)out_size; (void)ws_size;
  const float* x  = (const float*)d_in[0];
  const float* y  = (const float*)d_in[1];
  const float* Wq = (const float*)d_in[2];
  const float* bq = (const float*)d_in[3];
  const float* Wk = (const float*)d_in[4];
  const float* bk = (const float*)d_in[5];
  const float* Wv = (const float*)d_in[6];
  const float* bv = (const float*)d_in[7];

  char* ws = (char*)d_ws;
  __bf16* Wqb = (__bf16*)(ws + 0);        // 32 KB
  __bf16* Wkb = (__bf16*)(ws + 32768);    // 32 KB
  __bf16* Wvb = (__bf16*)(ws + 65536);    // 32 KB
  __bf16* Qb  = (__bf16*)(ws + 98304);                      // 8 MB
  __bf16* Kb  = (__bf16*)(ws + 98304 + 8388608);            // 2 MB
  __bf16* Vt  = (__bf16*)(ws + 98304 + 8388608 + 2097152);  // 2 MB

  k_convw<<<64, 256, 0, stream>>>(Wq, Wk, Wv, Wqb, Wkb, Wvb);
  k_proj_q<<<512, 128, 0, stream>>>(x, Wqb, bq, Qb);
  k_proj_kv<<<128, 128, 0, stream>>>(y, Wkb, bk, Wvb, bv, Kb, Vt);
  k_attn<<<512, 128, 0, stream>>>(x, Qb, Kb, Vt, (float*)d_out);
}
